// Model_60301340835919
// MI455X (gfx1250) — compile-verified, matrix operand order checked by reading, not run on hardware
//
#include <hip/hip_runtime.h>
#include <hip/hip_bf16.h>

typedef __attribute__((ext_vector_type(16))) __bf16 v16bf;
typedef __attribute__((ext_vector_type(8)))  float  v8f;
typedef unsigned short ushort_t;

#define B_  64
#define T_  512
#define E_  256
#define H_  256
#define G3_ 768
#define L_  9

__device__ __forceinline__ ushort_t f2bf(float f) {
    unsigned int u = __float_as_uint(f);
    unsigned int r = u + 0x7FFFu + ((u >> 16) & 1u);   // RNE
    return (ushort_t)(r >> 16);
}
__device__ __forceinline__ float bf2f(ushort_t u) {
    return __uint_as_float(((unsigned int)u) << 16);
}

// ---------------------------------------------------------------- kernel 0
// Convert f32 weights to bf16 into workspace.
__global__ void conv_weights(const float* __restrict__ Wihf, const float* __restrict__ Wihb,
                             const float* __restrict__ Whhf, const float* __restrict__ Whhb,
                             ushort_t* __restrict__ Wcat,
                             ushort_t* __restrict__ WhhF16, ushort_t* __restrict__ WhhB16) {
    int i = blockIdx.x * 256 + threadIdx.x;
    if (i < G3_ * E_) {
        Wcat[i]             = f2bf(Wihf[i]);
        Wcat[G3_ * E_ + i]  = f2bf(Wihb[i]);
        WhhF16[i]           = f2bf(Whhf[i]);
        WhhB16[i]           = f2bf(Whhb[i]);
    }
}

// ---------------------------------------------------------------- kernel 1
// Fused embedding gather + input projection GEMM (WMMA bf16, f32 accum).
// Output xg[dir][t][b][g] (f32), bias included. Grid: (M/64, 1536/64), 256 thr.
__global__ void __launch_bounds__(256)
embed_proj(const int* __restrict__ word_ids, const float* __restrict__ embed,
           const ushort_t* __restrict__ Wcat,
           const float* __restrict__ b_ih_f, const float* __restrict__ b_ih_b,
           float* __restrict__ xg) {
    __shared__ __attribute__((aligned(32))) ushort_t Abuf[64 * E_];
    __shared__ __attribute__((aligned(32))) ushort_t Bbuf[64 * E_];
    const int tid  = threadIdx.x;
    const int lane = tid & 31;
    const int wave = tid >> 5;           // 0..7
    const int m0   = blockIdx.x * 64;    // token base
    const int n0   = blockIdx.y * 64;    // gate-column base (0..1535)

    // Stage A: gather 64 embedding rows, f32 -> bf16.
    {
        int row = tid >> 2;
        int q   = (tid & 3) * 64;
        int id  = word_ids[m0 + row];
        const float4* src = (const float4*)(embed + (size_t)id * E_ + q);
        ushort_t* dst = Abuf + row * E_ + q;
#pragma unroll
        for (int k = 0; k < 16; ++k) {
            float4 v = src[k];
            dst[4 * k + 0] = f2bf(v.x); dst[4 * k + 1] = f2bf(v.y);
            dst[4 * k + 2] = f2bf(v.z); dst[4 * k + 3] = f2bf(v.w);
        }
    }
    // Stage B: copy 64 bf16 weight rows (these are columns of the GEMM B matrix).
    {
        const uint4* src = (const uint4*)(Wcat + (size_t)n0 * E_);
        uint4* dst = (uint4*)Bbuf;
#pragma unroll
        for (int it = 0; it < 8; ++it) dst[tid + it * 256] = src[tid + it * 256];
    }
    __syncthreads();

    const int mt   = wave & 3;
    const int ng   = wave >> 2;                 // 0..1 -> 2 n-tiles each
    const int lcol = lane & 15;
    const int koff = (lane >> 4) * 16;
    const int arow = mt * 16 + lcol;

    v8f acc0 = {}; v8f acc1 = {};
#pragma unroll
    for (int kc = 0; kc < 8; ++kc) {
        int k = kc * 32 + koff;
        v16bf a  = *(const v16bf*)(Abuf + arow * E_ + k);
        v16bf b0 = *(const v16bf*)(Bbuf + ((ng * 2 + 0) * 16 + lcol) * E_ + k);
        v16bf b1 = *(const v16bf*)(Bbuf + ((ng * 2 + 1) * 16 + lcol) * E_ + k);
        acc0 = __builtin_amdgcn_wmma_f32_16x16x32_bf16(false, a, false, b0, (short)0, acc0, false, false);
        acc1 = __builtin_amdgcn_wmma_f32_16x16x32_bf16(false, a, false, b1, (short)0, acc1, false, false);
    }

    const int ncol0 = n0 + (ng * 2 + 0) * 16 + lcol;
    const int dir   = (n0 >= G3_) ? 1 : 0;        // 64-col block never crosses 768
    const int g0    = ncol0 - dir * G3_;
    const int g1    = g0 + 16;
    const float bias0 = dir ? b_ih_b[g0] : b_ih_f[g0];
    const float bias1 = dir ? b_ih_b[g1] : b_ih_f[g1];

#pragma unroll
    for (int i = 0; i < 8; ++i) {
        int mrow = m0 + mt * 16 + i + ((lane >> 4) << 3);
        int b = mrow >> 9, t = mrow & 511;       // token = b*512 + t
        size_t base = ((size_t)dir * T_ * B_ + (size_t)t * B_ + b) * G3_;
        xg[base + g0] = acc0[i] + bias0;
        xg[base + g1] = acc1[i] + bias1;
    }
}

// ---------------------------------------------------------------- kernel 2
// Persistent GRU: one 1024-thread block per direction. Hidden state (bf16) and
// per-step gh (f32, 192KB) live in the 320KB WGP LDS. 48 WMMAs/wave/step.
__global__ void __launch_bounds__(1024)
gru_kernel(const ushort_t* __restrict__ WhhF16, const ushort_t* __restrict__ WhhB16,
           const float* __restrict__ bhhF, const float* __restrict__ bhhB,
           const float* __restrict__ xg, float* __restrict__ seq) {
    __shared__ __attribute__((aligned(32))) float    gh[B_ * G3_];    // 196608 B
    __shared__ __attribute__((aligned(32))) ushort_t hbuf[B_ * H_];   //  32768 B

    const int dir = blockIdx.x;
    const ushort_t* Whh = dir ? WhhB16 : WhhF16;
    const float*    bhh = dir ? bhhB : bhhF;
    const float*    xgd = xg + (size_t)dir * T_ * B_ * G3_;

    const int tid  = threadIdx.x;
    const int lane = tid & 31;
    const int wave = tid >> 5;              // 0..31
    const int mt   = wave & 3;              // 4 M tiles
    const int ng   = wave >> 2;             // 8 groups x 6 N tiles = 48
    const int lcol = lane & 15;
    const int koff = (lane >> 4) * 16;
    const int arow = mt * 16 + lcol;
    const int rsel = (lane >> 4) << 3;      // +8 rows for upper half-wave

    for (int i = tid; i < B_ * H_; i += 1024) hbuf[i] = 0;
    __syncthreads();

    for (int step = 0; step < T_; ++step) {
        const int t = dir ? (T_ - 1 - step) : step;

        // gh = h @ W_hh^T  (A from LDS, B streamed from L2-resident bf16 weights)
        v16bf afr[8];
#pragma unroll
        for (int kc = 0; kc < 8; ++kc)
            afr[kc] = *(const v16bf*)(hbuf + arow * H_ + kc * 32 + koff);

#pragma unroll
        for (int j = 0; j < 6; ++j) {
            const int nt = ng * 6 + j;
            const int n  = nt * 16 + lcol;
            v8f acc = {};
#pragma unroll
            for (int kc = 0; kc < 8; ++kc) {
                v16bf bfr = *(const v16bf*)(Whh + (size_t)n * H_ + kc * 32 + koff);
                acc = __builtin_amdgcn_wmma_f32_16x16x32_bf16(false, afr[kc], false, bfr,
                                                              (short)0, acc, false, false);
            }
#pragma unroll
            for (int i = 0; i < 8; ++i)
                gh[(mt * 16 + i + rsel) * G3_ + nt * 16 + lcol] = acc[i];
        }
        __syncthreads();

        // Gate math: 16384 (b,j) elements over 1024 threads.
        const float* gi = xgd + (size_t)t * B_ * G3_;
#pragma unroll
        for (int p = 0; p < 16; ++p) {
            int idx = tid + p * 1024;
            int b = idx >> 8, jj = idx & 255;
            float gir = gi[b * G3_ + jj];
            float giz = gi[b * G3_ + 256 + jj];
            float gin = gi[b * G3_ + 512 + jj];
            float ghr = gh[b * G3_ + jj]        + bhh[jj];
            float ghz = gh[b * G3_ + 256 + jj]  + bhh[256 + jj];
            float ghn = gh[b * G3_ + 512 + jj]  + bhh[512 + jj];
            float r  = 1.f / (1.f + __expf(-(gir + ghr)));
            float z  = 1.f / (1.f + __expf(-(giz + ghz)));
            float nn = tanhf(gin + r * ghn);
            float hold = bf2f(hbuf[b * H_ + jj]);
            float hnew = (1.f - z) * nn + z * hold;
            seq[((size_t)b * T_ + t) * (2 * H_) + dir * H_ + jj] = hnew;
            hbuf[b * H_ + jj] = f2bf(hnew);
        }
        __syncthreads();
    }
}

// ---------------------------------------------------------------- kernel 3
// Classifier (L=9) + per-token argmax. One thread per token.
__global__ void __launch_bounds__(256)
classifier(const float* __restrict__ seq, const float* __restrict__ clsW,
           const float* __restrict__ clsb, float* __restrict__ logits,
           float* __restrict__ outArg) {
    __shared__ float Ws[L_ * 2 * H_];
    __shared__ float bs[L_];
    int tid = threadIdx.x;
    for (int i = tid; i < L_ * 2 * H_; i += 256) Ws[i] = clsW[i];
    if (tid < L_) bs[tid] = clsb[tid];
    __syncthreads();

    int tok = blockIdx.x * 256 + tid;          // = b*T + t
    const float* row = seq + (size_t)tok * (2 * H_);
    float acc[L_];
#pragma unroll
    for (int l = 0; l < L_; ++l) acc[l] = bs[l];
    for (int k = 0; k < 2 * H_; ++k) {
        float s = row[k];
#pragma unroll
        for (int l = 0; l < L_; ++l) acc[l] += s * Ws[l * (2 * H_) + k];
    }
    int best = 0; float bv = acc[0];
#pragma unroll
    for (int l = 0; l < L_; ++l) {
        logits[(size_t)tok * L_ + l] = acc[l];
        if (acc[l] > bv) { bv = acc[l]; best = l; }
    }
    outArg[tok] = (float)best;
}

// ---------------------------------------------------------------- kernel 4
// CRF negative log-likelihood: one thread per batch row, block reduction.
__global__ void __launch_bounds__(64)
crf_kernel(const int* __restrict__ label_ids, const unsigned char* __restrict__ label_mask,
           const float* __restrict__ logits, const float* __restrict__ startT,
           const float* __restrict__ endT, const float* __restrict__ trans,
           float* __restrict__ out0) {
    __shared__ float part[B_];
    __shared__ float tr[L_ * L_], st[L_], en[L_];
    int tid = threadIdx.x;
    if (tid < L_ * L_) tr[tid] = trans[tid];
    if (tid < L_) { st[tid] = startT[tid]; en[tid] = endT[tid]; }
    __syncthreads();

    int b = tid;
    const int* tags = label_ids + b * T_;
    const unsigned char* mk = label_mask + b * T_;
    const float* em = logits + (size_t)b * T_ * L_;

    // Numerator
    float num = st[tags[0]] + em[tags[0]];
    int prev = tags[0];
    int cnt = mk[0] ? 1 : 0;
    for (int t = 1; t < T_; ++t) {
        int tg = tags[t];
        if (mk[t]) { num += tr[prev * L_ + tg] + em[t * L_ + tg]; ++cnt; }
        prev = tg;
    }
    num += en[tags[cnt - 1]];

    // Forward algorithm (log-space)
    float a[L_];
#pragma unroll
    for (int j = 0; j < L_; ++j) a[j] = st[j] + em[j];
    for (int t = 1; t < T_; ++t) {
        float nxt[L_];
#pragma unroll
        for (int j = 0; j < L_; ++j) {
            float mx = -3.4e38f;
#pragma unroll
            for (int i = 0; i < L_; ++i) mx = fmaxf(mx, a[i] + tr[i * L_ + j]);
            float s = 0.f;
#pragma unroll
            for (int i = 0; i < L_; ++i) s += __expf(a[i] + tr[i * L_ + j] - mx);
            nxt[j] = mx + __logf(s) + em[t * L_ + j];
        }
        if (mk[t]) {
#pragma unroll
            for (int j = 0; j < L_; ++j) a[j] = nxt[j];
        }
    }
    float mx = -3.4e38f;
#pragma unroll
    for (int j = 0; j < L_; ++j) mx = fmaxf(mx, a[j] + en[j]);
    float s = 0.f;
#pragma unroll
    for (int j = 0; j < L_; ++j) s += __expf(a[j] + en[j] - mx);
    float denom = mx + __logf(s);

    part[b] = num - denom;
    __syncthreads();
    if (tid == 0) {
        float tot = 0.f;
        for (int i = 0; i < B_; ++i) tot += part[i];
        out0[0] = -tot;
    }
}

// ---------------------------------------------------------------- launcher
extern "C" void kernel_launch(void* const* d_in, const int* in_sizes, int n_in,
                              void* d_out, int out_size, void* d_ws, size_t ws_size,
                              hipStream_t stream) {
    const int*   word_ids   = (const int*)d_in[0];
    const int*   label_ids  = (const int*)d_in[1];
    const unsigned char* label_mask = (const unsigned char*)d_in[2];
    const float* embed      = (const float*)d_in[3];
    const float* W_ih_f     = (const float*)d_in[4];
    const float* W_hh_f     = (const float*)d_in[5];
    const float* b_ih_f     = (const float*)d_in[6];
    const float* b_hh_f     = (const float*)d_in[7];
    const float* W_ih_b     = (const float*)d_in[8];
    const float* W_hh_b     = (const float*)d_in[9];
    const float* b_ih_b     = (const float*)d_in[10];
    const float* b_hh_b     = (const float*)d_in[11];
    const float* cls_W      = (const float*)d_in[12];
    const float* cls_b      = (const float*)d_in[13];
    const float* start_t    = (const float*)d_in[14];
    const float* end_t      = (const float*)d_in[15];
    const float* trans      = (const float*)d_in[16];

    char* ws = (char*)d_ws;
    size_t off = 0;
    ushort_t* Wcat   = (ushort_t*)(ws + off); off += (size_t)2 * G3_ * E_ * sizeof(ushort_t);
    ushort_t* WhhF16 = (ushort_t*)(ws + off); off += (size_t)G3_ * H_ * sizeof(ushort_t);
    ushort_t* WhhB16 = (ushort_t*)(ws + off); off += (size_t)G3_ * H_ * sizeof(ushort_t);
    float* xg     = (float*)(ws + off); off += (size_t)2 * T_ * B_ * G3_ * sizeof(float);
    float* seq    = (float*)(ws + off); off += (size_t)B_ * T_ * 2 * H_ * sizeof(float);
    float* logits = (float*)(ws + off); off += (size_t)B_ * T_ * L_ * sizeof(float);

    float* out_loss = (float*)d_out;
    float* out_arg  = (float*)d_out + 1;

    conv_weights<<<(G3_ * E_ + 255) / 256, 256, 0, stream>>>(
        W_ih_f, W_ih_b, W_hh_f, W_hh_b, Wcat, WhhF16, WhhB16);

    dim3 g1((B_ * T_) / 64, (2 * G3_) / 64);
    embed_proj<<<g1, 256, 0, stream>>>(word_ids, embed, Wcat, b_ih_f, b_ih_b, xg);

    gru_kernel<<<2, 1024, 0, stream>>>(WhhF16, WhhB16, b_hh_f, b_hh_b, xg, seq);

    classifier<<<(B_ * T_) / 256, 256, 0, stream>>>(seq, cls_W, cls_b, logits, out_arg);

    crf_kernel<<<1, 64, 0, stream>>>(label_ids, label_mask, logits,
                                     start_t, end_t, trans, out_loss);
}
